// GraphSAGE_63015760166968
// MI455X (gfx1250) — matrix operand order repaired and checked by reading, not compiled, not run on hardware
//
#include <hip/hip_runtime.h>
#include <hip/hip_bf16.h>

typedef __attribute__((ext_vector_type(16))) _Float16 v16h;
typedef __attribute__((ext_vector_type(8)))  float    v8f;
typedef __attribute__((ext_vector_type(4)))  int      v4i;

union Frag16 { unsigned int u[8]; v16h v; };

static __device__ __forceinline__ unsigned int pack2h(float f0, float f1) {
    _Float16 h0 = (_Float16)f0, h1 = (_Float16)f1;
    return (unsigned int)__builtin_bit_cast(unsigned short, h0) |
           ((unsigned int)__builtin_bit_cast(unsigned short, h1) << 16);
}

// 32-bit LDS byte offset of a generic pointer that points into LDS
static __device__ __forceinline__ unsigned lds_off(const void* p) {
    return (unsigned)(uintptr_t)(__attribute__((address_space(3))) const void*)(p);
}

// async global -> LDS, 16 bytes per lane (GLOBAL_LOAD_ASYNC_TO_LDS_B128, ASYNCcnt)
static __device__ __forceinline__ void async_copy_b128(unsigned ldsByteOff, const void* gaddr) {
#if __has_builtin(__builtin_amdgcn_global_load_async_to_lds_b128)
    __builtin_amdgcn_global_load_async_to_lds_b128(
        (__attribute__((address_space(1))) v4i*)(uintptr_t)gaddr,
        (__attribute__((address_space(3))) v4i*)(uintptr_t)ldsByteOff,
        0, 0);
#else
    asm volatile("global_load_async_to_lds_b128 %0, %1, off"
                 :: "v"(ldsByteOff), "v"(gaddr) : "memory");
#endif
}

static __device__ __forceinline__ void wait_asynccnt0() {
#if __has_builtin(__builtin_amdgcn_s_wait_asynccnt)
    __builtin_amdgcn_s_wait_asynccnt(0);
#else
    asm volatile("s_wait_asynccnt 0x0" ::: "memory");
#endif
}

// ---------------- edge degree count: cnt[dst] += 1 ----------------
__global__ void __launch_bounds__(256)
count_edges(const int* __restrict__ dst, float* __restrict__ cnt, int nE) {
    int t = blockIdx.x * blockDim.x + threadIdx.x;
    if (t < nE) atomicAdd(&cnt[dst[t]], 1.0f);
}

// ---------------- scatter-add: agg[dst] += x[src], float4 per thread ----------------
// vshift = log2(dim/4); dim is a power of two (128 or 256)
__global__ void __launch_bounds__(256)
scatter_feat4(const float* __restrict__ x, const int* __restrict__ src,
              const int* __restrict__ dst, float* __restrict__ agg,
              unsigned total, int dim, int vshift) {
    unsigned t = blockIdx.x * blockDim.x + threadIdx.x;
    if (t >= total) return;
    unsigned e = t >> vshift;
    unsigned c = (t & (((unsigned)1 << vshift) - 1u)) << 2;
    int s = src[e], d = dst[e];
    const float4 v = *reinterpret_cast<const float4*>(x + (size_t)s * dim + c);
    float* p = agg + (size_t)d * dim + c;
    atomicAdd(p + 0, v.x); atomicAdd(p + 1, v.y);
    atomicAdd(p + 2, v.z); atomicAdd(p + 3, v.w);
}

// ---------------- mean = agg / max(cnt, 1) (in place); shift = log2(dim) ----------------
__global__ void __launch_bounds__(256)
mean_div(float* __restrict__ agg, const float* __restrict__ cnt, unsigned total, int shift) {
    unsigned t = blockIdx.x * blockDim.x + threadIdx.x;
    if (t >= total) return;
    float c = cnt[t >> shift];
    agg[t] = agg[t] / fmaxf(c, 1.0f);
}

// ---------------- dual-A concatenated-K GEMM with WMMA f16 ----------------
// out[M x N] = A1[M x KD] @ B1[KD x N] + A2[M x KD] @ B2[KD x N] + bias (+ ReLU)
// Block = 256 threads = 8 waves tiled 2(M) x 4(N); block tile = 32 rows x 64 cols.
// K-step 64: A chunk staged f32 via async b128 copies, 2 chained WMMAs / iter.
// N, KD compile-time -> immediate-offset loads, fully unrolled K loop.
#define ASTRIDE 68   // dwords; 272B row stride: 16B-aligned, conflict-free lane stride
template<int N, int KD, bool RELU>
__global__ void __launch_bounds__(256)
sage_gemm(const float* __restrict__ A1, const float* __restrict__ A2,
          const float* __restrict__ B1, const float* __restrict__ B2,
          const float* __restrict__ bias, float* __restrict__ out,
          int Mtot)
{
    __shared__ float ldsA[32][ASTRIDE];   // 32 rows x 64 k, f32

    const int tid   = threadIdx.x;
    const int lane  = tid & 31;
    const int wave  = tid >> 5;
    const int wm    = wave & 1;       // M sub-tile (0..1)
    const int wn    = wave >> 1;      // N sub-tile (0..3)
    const int lm    = lane & 15;
    const int lhalf = lane >> 4;
    const int rowbase = blockIdx.x * 32;
    const int n = blockIdx.y * 64 + wn * 16 + lm;
    const int arow = wm * 16 + lm;

    // per-thread async-copy coordinates: 512 x b128 transfers (32 rows x 16 chunks), 2/thread
    const int cr0 = tid >> 4,        ci0 = (tid & 15);        // rows 0..15
    const int cr1 = 16 + (tid >> 4), ci1 = (tid & 15);        // rows 16..31

    v8f acc = {};
    #pragma unroll
    for (int ks = 0; ks < 2 * KD; ks += 64) {
        const float* Ap; const float* Bp; int k0;
        if (ks < KD) { Ap = A1; Bp = B1; k0 = ks; }
        else         { Ap = A2; Bp = B2; k0 = ks - KD; }

        // stage A 32x64 f32 chunk into LDS with async b128 copies
        __syncthreads();
        {
            int rr0 = rowbase + cr0; if (rr0 >= Mtot) rr0 = Mtot - 1;
            int rr1 = rowbase + cr1; if (rr1 >= Mtot) rr1 = Mtot - 1;
            async_copy_b128(lds_off(&ldsA[cr0][ci0 * 4]),
                            Ap + (size_t)rr0 * KD + k0 + ci0 * 4);
            async_copy_b128(lds_off(&ldsA[cr1][ci1 * 4]),
                            Ap + (size_t)rr1 * KD + k0 + ci1 * 4);
        }
        wait_asynccnt0();
        __syncthreads();

        // per-lane weight base: k-dependent offsets are compile-time immediates
        const float* Bcol = Bp + (size_t)k0 * N + (size_t)lhalf * 16 * N + n;

        #pragma unroll
        for (int ksub = 0; ksub < 64; ksub += 32) {
            // A fragment in contiguous k order (k = 16*lhalf + s), same as B; f32->f16 at read time
            Frag16 a;
            #pragma unroll
            for (int v = 0; v < 8; ++v) {
                int k = ksub + lhalf * 16 + v * 2;
                a.u[v] = pack2h(ldsA[arow][k], ldsA[arow][k + 1]);
            }

            // B fragment from global weights (f32 -> f16): lane n fixed, k varies
            Frag16 b;
            #pragma unroll
            for (int v = 0; v < 8; ++v) {
                int off = (ksub + 2 * v) * N;       // compile-time immediate
                float f0 = Bcol[off];
                float f1 = Bcol[off + N];
                b.u[v] = pack2h(f0, f1);
            }

            acc = __builtin_amdgcn_wmma_f32_16x16x32_f16(
                /*neg_a=*/false, a.v, /*neg_b=*/false, b.v,
                /*c_mod=*/(short)0, acc, /*reuse_a=*/false, /*reuse_b=*/false);
        }
    }

    const float bn = bias[n];
    float* orow = out + (size_t)(rowbase + wm * 16 + (lhalf ? 8 : 0)) * N + n;
    const int row0 = rowbase + wm * 16 + (lhalf ? 8 : 0);
    #pragma unroll
    for (int r = 0; r < 8; ++r) {
        if (row0 + r < Mtot) {
            float val = acc[r] + bn;
            if (RELU) val = fmaxf(val, 0.0f);
            orow[r * N] = val;                       // compile-time immediate stride
        }
    }
}

// ---------------- log_softmax over 64 cols, one wave32 per row ----------------
__global__ void __launch_bounds__(256)
logsoftmax64(const float* __restrict__ z, float* __restrict__ out, int M) {
    int warp = (blockIdx.x * blockDim.x + threadIdx.x) >> 5;
    int lane = threadIdx.x & 31;
    if (warp >= M) return;
    const float* zr = z + (size_t)warp * 64;
    float v0 = zr[lane], v1 = zr[lane + 32];
    float m = fmaxf(v0, v1);
    #pragma unroll
    for (int off = 16; off; off >>= 1) m = fmaxf(m, __shfl_xor(m, off, 32));
    float s = __expf(v0 - m) + __expf(v1 - m);
    #pragma unroll
    for (int off = 16; off; off >>= 1) s += __shfl_xor(s, off, 32);
    float l = __logf(s);
    float* orow = out + (size_t)warp * 64;
    orow[lane]      = v0 - m - l;
    orow[lane + 32] = v1 - m - l;
}

extern "C" void kernel_launch(void* const* d_in, const int* in_sizes, int n_in,
                              void* d_out, int out_size, void* d_ws, size_t ws_size,
                              hipStream_t stream) {
    const float* x   = (const float*)d_in[0];
    const int*   ei  = (const int*)  d_in[1];
    const float* W1l = (const float*)d_in[2];
    const float* b1  = (const float*)d_in[3];
    const float* W1r = (const float*)d_in[4];
    const float* W2l = (const float*)d_in[5];
    const float* b2  = (const float*)d_in[6];
    const float* W2r = (const float*)d_in[7];

    const int nE = in_sizes[1] / 2;
    const int M  = in_sizes[0] / 128;       // 100000
    const int* src = ei;
    const int* dst = ei + nE;

    // workspace layout (256B aligned slabs)
    char* ws = (char*)d_ws;
    size_t o = 0;
    auto alloc = [&](size_t bytes) {
        void* p = ws + o;
        o = (o + bytes + 255) & ~(size_t)255;
        return p;
    };
    float* cnt  = (float*)alloc((size_t)M * 4);           // degrees
    float* agg1 = (float*)alloc((size_t)M * 128 * 4);     // sum/mean of x
    float* h    = (float*)alloc((size_t)M * 256 * 4);     // layer-1 activations
    float* agg2 = (float*)alloc((size_t)M * 256 * 4);     // sum/mean of h
    float* z    = (float*)alloc((size_t)M * 64 * 4);      // layer-2 pre-softmax

    (void)hipMemsetAsync(cnt,  0, (size_t)M * 4,         stream);
    (void)hipMemsetAsync(agg1, 0, (size_t)M * 128 * 4,   stream);
    (void)hipMemsetAsync(agg2, 0, (size_t)M * 256 * 4,   stream);

    count_edges<<<(nE + 255) / 256, 256, 0, stream>>>(dst, cnt, nE);

    {   // layer-1 aggregation: 32 float4 threads per edge
        unsigned total = (unsigned)nE * 32u;
        scatter_feat4<<<(total + 255) / 256, 256, 0, stream>>>(x, src, dst, agg1, total, 128, 5);
    }
    {
        unsigned total = (unsigned)M * 128u;
        mean_div<<<(total + 255) / 256, 256, 0, stream>>>(agg1, cnt, total, 7);
    }
    {   // h = relu(mean @ W1_l + b1 + x @ W1_r)
        dim3 grid((M + 31) / 32, 256 / 64);
        sage_gemm<256, 128, true><<<grid, 256, 0, stream>>>(agg1, x, W1l, W1r, b1, h, M);
    }
    {   // layer-2 aggregation: 64 float4 threads per edge
        unsigned total = (unsigned)nE * 64u;
        scatter_feat4<<<(total + 255) / 256, 256, 0, stream>>>(h, src, dst, agg2, total, 256, 6);
    }
    {
        unsigned total = (unsigned)M * 256u;
        mean_div<<<(total + 255) / 256, 256, 0, stream>>>(agg2, cnt, total, 8);
    }
    {   // z = mean2 @ W2_l + b2 + h @ W2_r
        dim3 grid((M + 31) / 32, 1);
        sage_gemm<64, 256, false><<<grid, 256, 0, stream>>>(agg2, h, W2l, W2r, b2, z, M);
    }
    {   // out = log_softmax(z, axis=1)
        unsigned threads = (unsigned)M * 32u;
        logsoftmax64<<<(threads + 255) / 256, 256, 0, stream>>>(z, (float*)d_out, M);
    }
}
